// Attention_1039382086273
// MI455X (gfx1250) — matrix operand order repaired and checked by reading, not compiled, not run on hardware
//
#include <hip/hip_runtime.h>
#include <hip/hip_bf16.h>

typedef __attribute__((ext_vector_type(16))) __bf16 v16bf;
typedef __attribute__((ext_vector_type(8)))  __bf16 v8bf;
typedef __attribute__((ext_vector_type(4)))  __bf16 v4bf;
typedef __attribute__((ext_vector_type(8)))  float  v8f;

#define DIM   1024
#define SEQ   2048
#define BATCH 4
#define HEADS 16
#define DH    64
#define ROWS  (BATCH*SEQ)   // 8192
#define EPS   1e-5f

#ifndef __has_builtin
#define __has_builtin(x) 0
#endif

static __device__ __forceinline__ v16bf cat16(v8bf lo, v8bf hi) {
  union U { struct { v8bf lo, hi; } s; v16bf v; } u;
  u.s.lo = lo; u.s.hi = hi; return u.v;
}

// 16-byte async copy global -> LDS (ASYNCcnt-tracked, no VGPR round trip).
// GV addressing mode: vdst = LDS byte offset (VGPR), vaddr = 64-bit address.
static __device__ __forceinline__ void async_copy16(const __bf16* g, __bf16* l) {
#if defined(__AMDGCN__)
  unsigned loff = (unsigned)(__UINTPTR_TYPE__)(__attribute__((address_space(3))) __bf16*)l;
  asm volatile("global_load_async_to_lds_b128 %0, %1, off"
               :: "v"(loff), "v"(g) : "memory");
#else
  (void)g; (void)l;
#endif
}

static __device__ __forceinline__ void async_wait0() {
#if defined(__AMDGCN__)
#if __has_builtin(__builtin_amdgcn_s_wait_asynccnt)
  __builtin_amdgcn_s_wait_asynccnt(0);
#else
  asm volatile("s_wait_asynccnt 0x0" ::: "memory");
#endif
#endif
}

// ---------------------------------------------------------------------------
// 1) LayerNorm: fp32 [ROWS, DIM] -> bf16 [ROWS, DIM]
// ---------------------------------------------------------------------------
__global__ __launch_bounds__(256) void ln_kernel(const float* __restrict__ x,
                                                 const float* __restrict__ g,
                                                 const float* __restrict__ b,
                                                 __bf16* __restrict__ out) {
  const int row = blockIdx.x;
  const int tid = threadIdx.x;
  const float* xr = x + (size_t)row * DIM;
  float4 v = *(const float4*)(xr + tid * 4);
  float s = v.x + v.y + v.z + v.w;
  float q = v.x*v.x + v.y*v.y + v.z*v.z + v.w*v.w;
  #pragma unroll
  for (int m = 16; m >= 1; m >>= 1) { s += __shfl_xor(s, m); q += __shfl_xor(q, m); }
  __shared__ float rs[8], rq[8];
  const int w = tid >> 5, l = tid & 31;
  if (l == 0) { rs[w] = s; rq[w] = q; }
  __syncthreads();
  s = 0.f; q = 0.f;
  #pragma unroll
  for (int i = 0; i < 8; i++) { s += rs[i]; q += rq[i]; }
  const float mean = s * (1.0f / DIM);
  const float var  = q * (1.0f / DIM) - mean * mean;
  const float rstd = rsqrtf(var + EPS);
  float4 gv = *(const float4*)(g + tid * 4);
  float4 bv = *(const float4*)(b + tid * 4);
  v4bf o;
  o[0] = (__bf16)((v.x - mean) * rstd * gv.x + bv.x);
  o[1] = (__bf16)((v.y - mean) * rstd * gv.y + bv.y);
  o[2] = (__bf16)((v.z - mean) * rstd * gv.z + bv.z);
  o[3] = (__bf16)((v.w - mean) * rstd * gv.w + bv.w);
  *(v4bf*)(out + (size_t)row * DIM + tid * 4) = o;
}

// ---------------------------------------------------------------------------
// 2) fp32 [K][N] -> bf16 transposed [N][K]  (weights, one-time)
// ---------------------------------------------------------------------------
__global__ __launch_bounds__(256) void cvt_t_kernel(const float* __restrict__ in,
                                                    __bf16* __restrict__ out,
                                                    int N, int K) {
  const int idx = blockIdx.x * 256 + threadIdx.x;
  if (idx < N * K) {
    const int n = idx / K, kk = idx - n * K;
    out[(size_t)n * K + kk] = (__bf16)in[(size_t)kk * N + n];
  }
}

// ---------------------------------------------------------------------------
// 3/5) bf16 WMMA GEMM: C[M,N] = A[M,K] * Bt[N,K]^T
//      Block 64x128, BK=32, double-buffered async-to-LDS staging.
//      MODE 1: scatter into q[b,h,n,d], k[b,h,n,d], vT[b,h,d,n] (bf16)
//      MODE 2: f32 + bias row-major
// ---------------------------------------------------------------------------
template <int MODE>
__global__ __launch_bounds__(256) void gemm_bf16(const __bf16* __restrict__ A,
                                                 const __bf16* __restrict__ Bt,
                                                 __bf16* __restrict__ qo,
                                                 __bf16* __restrict__ ko,
                                                 __bf16* __restrict__ vto,
                                                 float* __restrict__ fo,
                                                 const float* __restrict__ bias,
                                                 int N, int K) {
  __shared__ __align__(16) __bf16 As[2][64][40];    // [m][k], stride 80B
  __shared__ __align__(16) __bf16 Bs[2][128][40];   // [n][k], stride 80B
  const int tid = threadIdx.x;
  const int mBase = blockIdx.y * 64, nBase = blockIdx.x * 128;
  const int wave = tid >> 5, lane = tid & 31, lr = lane & 15, lh = lane >> 4;
  const int wm = (wave & 1) * 32, wn = (wave >> 1) * 32;

  v8f acc[2][2] = {};
  const int arow = tid >> 2, acol = (tid & 3) * 8;   // A tile 64x32
  const int brow = tid >> 1, bcol = (tid & 1) * 16;  // B tile 128x32

  const __bf16* aptr = A  + (size_t)(mBase + arow) * K + acol;
  const __bf16* bptr = Bt + (size_t)(nBase + brow) * K + bcol;

  auto issue = [&](int buf, int k0) {
    async_copy16(aptr + k0,     &As[buf][arow][acol]);
    async_copy16(bptr + k0,     &Bs[buf][brow][bcol]);
    async_copy16(bptr + k0 + 8, &Bs[buf][brow][bcol + 8]);
  };

  issue(0, 0);
  async_wait0();
  __syncthreads();

  const int nk = K >> 5;
  for (int t = 0; t < nk; ++t) {
    const int cur = t & 1;
    if (t + 1 < nk) issue(cur ^ 1, (t + 1) << 5);
    if (t + 2 < nk) {                       // global_prefetch_b8
      __builtin_prefetch(aptr + ((t + 2) << 5), 0, 1);
      __builtin_prefetch(bptr + ((t + 2) << 5), 0, 1);
    }
    v16bf af[2], bfm[2];
    #pragma unroll
    for (int r = 0; r < 2; r++)
      af[r] = cat16(*(const v8bf*)&As[cur][wm + r * 16 + lr][lh * 8],
                    *(const v8bf*)&As[cur][wm + r * 16 + lr][16 + lh * 8]);
    #pragma unroll
    for (int c = 0; c < 2; c++)
      bfm[c] = cat16(*(const v8bf*)&Bs[cur][wn + c * 16 + lr][lh * 8],
                     *(const v8bf*)&Bs[cur][wn + c * 16 + lr][16 + lh * 8]);
    #pragma unroll
    for (int r = 0; r < 2; r++)
      #pragma unroll
      for (int c = 0; c < 2; c++)
        acc[r][c] = __builtin_amdgcn_wmma_f32_16x16x32_bf16(
            false, af[r], false, bfm[c], (short)0, acc[r][c], false, false);
    async_wait0();
    __syncthreads();
  }

  // Epilogue. C layout: VGPR e holds row (lh*8 + e); col = lr per 16x16 tile.
  #pragma unroll
  for (int r = 0; r < 2; r++)
    #pragma unroll
    for (int c = 0; c < 2; c++)
      #pragma unroll
      for (int e = 0; e < 8; e++) {
        const int m   = mBase + wm + r * 16 + lh * 8 + e;
        const int col = nBase + wn + c * 16 + lr;
        const float val = acc[r][c][e];
        if (MODE == 2) {
          fo[(size_t)m * N + col] = val + bias[col];
        } else {
          const int bz = m >> 11, nrow = m & (SEQ - 1);
          const int sect = col >> 10, cc = col & (DIM - 1);
          const int h = cc >> 6, d = cc & (DH - 1);
          const size_t bh = (size_t)bz * HEADS + h;
          if (sect == 0)      qo[(bh * SEQ + nrow) * DH + d] = (__bf16)val;
          else if (sect == 1) ko[(bh * SEQ + nrow) * DH + d] = (__bf16)val;
          else                vto[(bh * DH + d) * SEQ + nrow] = (__bf16)val;
        }
      }
}

// ---------------------------------------------------------------------------
// 4) Flash attention. q,k: bf16 [b,h,n,dh]; vT: bf16 [b,h,dh,n].
//    Block = 128 query rows of one (b,h); 8 waves x 16 rows.
//    Double-buffered async K/Vt tiles (32 keys per step).
// ---------------------------------------------------------------------------
__global__ __launch_bounds__(256) void flash_kernel(const __bf16* __restrict__ q,
                                                    const __bf16* __restrict__ k,
                                                    const __bf16* __restrict__ vt,
                                                    __bf16* __restrict__ outp) {
  __shared__ __align__(16) __bf16 Ks[2][32][72];    // [key][d], stride 144B
  __shared__ __align__(16) __bf16 Vs[2][64][40];    // [d][key], stride 80B
  __shared__ __align__(16) __bf16 Ps[8][16][48];    // per-wave P, stride 96B
  const int tid = threadIdx.x, wave = tid >> 5, lane = tid & 31;
  const int lr = lane & 15, lh = lane >> 4;
  const int qb = blockIdx.x, h = blockIdx.y, bz = blockIdx.z;
  const size_t bh = (size_t)bz * HEADS + h;
  const int qrow0 = qb * 128 + wave * 16;

  // Q fragments (16x32 A-layout) in registers
  const __bf16* qptr = q + (bh * SEQ + qrow0 + lr) * DH;
  v16bf qf[2];
  #pragma unroll
  for (int kb = 0; kb < 2; kb++)
    qf[kb] = cat16(*(const v8bf*)(qptr + kb * 32 + lh * 8),
                   *(const v8bf*)(qptr + kb * 32 + 16 + lh * 8));

  v8f o[4] = {};
  float mrow[8], lrow[8];
  #pragma unroll
  for (int r = 0; r < 8; r++) { mrow[r] = -1e30f; lrow[r] = 0.f; }
  const float scale = 0.125f;  // 1/sqrt(64)

  const __bf16* kbase = k  + bh * SEQ * DH;
  const __bf16* vbase = vt + bh * DH * SEQ;
  const int kr = tid >> 3, kc = (tid & 7) * 8;   // K tile 32x64
  const int vr = tid >> 2, vc = (tid & 3) * 8;   // Vt tile 64x32

  auto issue = [&](int buf, int kv0) {
    async_copy16(kbase + (size_t)(kv0 + kr) * DH + kc, &Ks[buf][kr][kc]);
    async_copy16(vbase + (size_t)vr * SEQ + kv0 + vc,  &Vs[buf][vr][vc]);
  };

  issue(0, 0);
  async_wait0();
  __syncthreads();

  for (int t = 0; t < SEQ / 32; ++t) {
    const int cur = t & 1;
    if (t + 1 < SEQ / 32) issue(cur ^ 1, (t + 1) * 32);

    // Hoist ALL K fragments so ds_loads overlap the WMMA group.
    v16bf kf[2][2];
    #pragma unroll
    for (int cb = 0; cb < 2; cb++) {
      kf[cb][0] = cat16(*(const v8bf*)&Ks[cur][cb * 16 + lr][lh * 8],
                        *(const v8bf*)&Ks[cur][cb * 16 + lr][16 + lh * 8]);
      kf[cb][1] = cat16(*(const v8bf*)&Ks[cur][cb * 16 + lr][32 + lh * 8],
                        *(const v8bf*)&Ks[cur][cb * 16 + lr][32 + 16 + lh * 8]);
    }
    // S = Q K^T (two 16-key tiles, 2 chained WMMAs each over d=64)
    v8f s[2];
    #pragma unroll
    for (int cb = 0; cb < 2; cb++) {
      v8f z = {};
      z = __builtin_amdgcn_wmma_f32_16x16x32_bf16(false, qf[0], false, kf[cb][0], (short)0, z, false, false);
      z = __builtin_amdgcn_wmma_f32_16x16x32_bf16(false, qf[1], false, kf[cb][1], (short)0, z, false, false);
      s[cb] = z;
    }

    // Online softmax (rows across 16-lane halves per C layout)
    #pragma unroll
    for (int r = 0; r < 8; r++) {
      const float a0 = s[0][r] * scale, a1 = s[1][r] * scale;
      float mx = fmaxf(a0, a1);
      #pragma unroll
      for (int d = 8; d >= 1; d >>= 1) mx = fmaxf(mx, __shfl_xor(mx, d));
      const float nm = fmaxf(mrow[r], mx);
      const float corr = __expf(mrow[r] - nm);
      mrow[r] = nm;
      const float p0 = __expf(a0 - nm), p1 = __expf(a1 - nm);
      float ps = p0 + p1;
      #pragma unroll
      for (int d = 8; d >= 1; d >>= 1) ps += __shfl_xor(ps, d);
      lrow[r] = lrow[r] * corr + ps;
      #pragma unroll
      for (int db = 0; db < 4; db++) o[db][r] *= corr;
      Ps[wave][lh * 8 + r][lr]      = (__bf16)p0;
      Ps[wave][lh * 8 + r][16 + lr] = (__bf16)p1;
    }
    asm volatile("s_wait_dscnt 0" ::: "memory");  // per-wave LDS RAW fence

    // O += P * V : hoist P and all 4 V fragments before the WMMA group.
    v16bf pf = cat16(*(const v8bf*)&Ps[wave][lr][lh * 8],
                     *(const v8bf*)&Ps[wave][lr][16 + lh * 8]);
    v16bf vf[4];
    #pragma unroll
    for (int db = 0; db < 4; db++)
      vf[db] = cat16(*(const v8bf*)&Vs[cur][db * 16 + lr][lh * 8],
                     *(const v8bf*)&Vs[cur][db * 16 + lr][16 + lh * 8]);
    #pragma unroll
    for (int db = 0; db < 4; db++)
      o[db] = __builtin_amdgcn_wmma_f32_16x16x32_bf16(false, pf, false, vf[db], (short)0, o[db], false, false);
    async_wait0();
    __syncthreads();
  }

  float inv[8];
  #pragma unroll
  for (int r = 0; r < 8; r++) inv[r] = 1.0f / lrow[r];
  #pragma unroll
  for (int db = 0; db < 4; db++)
    #pragma unroll
    for (int r = 0; r < 8; r++) {
      const size_t m = (size_t)bz * SEQ + qrow0 + lh * 8 + r;
      const int col  = h * DH + db * 16 + lr;
      outp[m * DIM + col] = (__bf16)(o[db][r] * inv[r]);
    }
}

// ---------------------------------------------------------------------------
extern "C" void kernel_launch(void* const* d_in, const int* in_sizes, int n_in,
                              void* d_out, int out_size, void* d_ws, size_t ws_size,
                              hipStream_t stream) {
  const float* x    = (const float*)d_in[0];
  const float* g    = (const float*)d_in[1];
  const float* be   = (const float*)d_in[2];
  const float* wqkv = (const float*)d_in[3];
  const float* wout = (const float*)d_in[4];
  const float* bout = (const float*)d_in[5];
  float* out = (float*)d_out;

  char* ws = (char*)d_ws;
  const size_t MB = 1024 * 1024;
  __bf16* xn   = (__bf16*)(ws);              // 16 MB  [ROWS][DIM]
  __bf16* wqT  = (__bf16*)(ws + 16 * MB);    //  6 MB  [3*DIM][DIM]
  __bf16* woT  = (__bf16*)(ws + 24 * MB);    //  2 MB  [DIM][DIM]
  __bf16* qb   = (__bf16*)(ws + 32 * MB);    // 16 MB  [b,h,n,dh]
  __bf16* kb   = (__bf16*)(ws + 48 * MB);    // 16 MB  [b,h,n,dh]
  __bf16* vtb  = (__bf16*)(ws + 64 * MB);    // 16 MB  [b,h,dh,n]
  __bf16* attn = (__bf16*)(ws + 80 * MB);    // 16 MB  [ROWS][DIM]

  ln_kernel<<<ROWS, 256, 0, stream>>>(x, g, be, xn);
  cvt_t_kernel<<<(3 * DIM * DIM + 255) / 256, 256, 0, stream>>>(wqkv, wqT, 3 * DIM, DIM);
  cvt_t_kernel<<<(DIM * DIM + 255) / 256, 256, 0, stream>>>(wout, woT, DIM, DIM);
  gemm_bf16<1><<<dim3(3 * DIM / 128, ROWS / 64), 256, 0, stream>>>(
      xn, wqT, qb, kb, vtb, nullptr, nullptr, 3 * DIM, DIM);
  flash_kernel<<<dim3(SEQ / 128, HEADS, BATCH), 256, 0, stream>>>(qb, kb, vtb, attn);
  gemm_bf16<2><<<dim3(DIM / 128, ROWS / 64), 256, 0, stream>>>(
      attn, woT, nullptr, nullptr, nullptr, out, bout, DIM, DIM);
}
